// ImprovedMultiHeadDotProductAttention_25598005084239
// MI455X (gfx1250) — matrix operand order repaired and checked by reading, not compiled, not run on hardware
//
#include <hip/hip_runtime.h>
#include <hip/hip_bf16.h>

typedef __attribute__((ext_vector_type(16))) _Float16 v16h;
typedef __attribute__((ext_vector_type(8)))  _Float16 v8h;
typedef __attribute__((ext_vector_type(8)))  float    v8f;
typedef __attribute__((ext_vector_type(4)))  float    v4f;

union V16U { v16h v; v8h h[2]; };

#define BB 2
#define QQ 2048
#define KK 2048
#define DD 1024
#define HH 16
#define HD 64
#define LN_EPS 1e-6f
#define BIG_NEG -3.402823466e38f

// ---- WMMA fragment swizzles (so every lane reads its operand contiguously) ----
// A-matrix 16x32 f16: lane L = g*16+m (g=K-half), half j -> K=g*16+j
__device__ __forceinline__ int a_off(int m, int k) {        // k in [0,32)
  return ((((k >> 4) & 1) * 16 + m) << 4) + (k & 15);
}
// B-matrix 32x16 f16: lane L = (k>>4)*16+n, half j -> K=(L>>4)*16+j
__device__ __forceinline__ int b_off(int k, int n) {        // k in [0,32), n in [0,16)
  return ((((k >> 4) & 1) * 16 + n) << 4) + (k & 15);
}

__device__ __forceinline__ v8f wmma_f16(v16h a, v16h b, v8f c) {
  return __builtin_amdgcn_wmma_f32_16x16x32_f16(false, a, false, b, (short)0, c,
                                                false, false);
}

// =====================================================================
// Kernel 1: QKV projection (+head bias) + QK layernorm (+1/sqrt(HD) on Q)
//   grid = (B*Q/64, H, 3[q,k,v]); block = 128 (4 waves)
//   writes q,k as [b,h,pos,HD] f16; v transposed as [b,h,HD,pos] f16
// =====================================================================
__global__ __launch_bounds__(128) void qkv_proj_kernel(
    const float* __restrict__ x,
    const float* __restrict__ wq, const float* __restrict__ wk, const float* __restrict__ wv,
    const float* __restrict__ bq, const float* __restrict__ bk, const float* __restrict__ bv,
    const float* __restrict__ qs, const float* __restrict__ ksc,
    _Float16* __restrict__ qo, _Float16* __restrict__ ko, _Float16* __restrict__ vt)
{
  __shared__ _Float16 sA[4 * 512];   // 4 M-frags, swizzled
  __shared__ _Float16 sB[4 * 512];   // 4 N-frags, swizzled
  __shared__ _Float16 sC[64 * 64];   // epilogue staging

  const int tid  = threadIdx.x;
  const int lane = tid & 31, wave = tid >> 5;
  const int g = lane >> 4, ln = lane & 15;
  const int which = blockIdx.z;
  const int h     = blockIdx.y;
  const int row0  = blockIdx.x * 64;          // row over flattened [B*Q]
  const int b     = row0 >> 11;               // / Q
  const int pos0  = row0 & (QQ - 1);

  const float* w       = (which == 0) ? wq : (which == 1) ? wk : wv;
  const float* bias_hd = (which == 0) ? bq : (which == 1) ? bk : bv;

  // staging assignments
  const int rowA = tid >> 1, cA = (tid & 1) * 16;     // A: 64 rows x 32 k
  const int nB   = tid & 63, kB = (tid >> 6) * 16;    // B: 32 k x 64 n (col/thread)

  float ar[16], br[16];
  auto load_gmem = [&](int kk) {
    const float* ap = x + (size_t)(row0 + rowA) * DD + kk * 32 + cA;
#pragma unroll
    for (int i = 0; i < 4; ++i) {
      v4f t = *(const v4f*)(ap + i * 4);
      ar[i*4+0] = t[0]; ar[i*4+1] = t[1]; ar[i*4+2] = t[2]; ar[i*4+3] = t[3];
    }
    const float* bp = w + (size_t)(kk * 32 + kB) * DD + h * HD + nB;
#pragma unroll
    for (int i = 0; i < 16; ++i) br[i] = bp[(size_t)i * DD];  // coalesced across threads
  };

  v8f zero8 = {};
  v8f acc[4] = { zero8, zero8, zero8, zero8 };

  load_gmem(0);
  for (int kk = 0; kk < DD / 32; ++kk) {
    __syncthreads();
    {
      int mf = rowA >> 4, ml = rowA & 15;
      v8h a0, a1;
#pragma unroll
      for (int i = 0; i < 8; ++i) { a0[i] = (_Float16)ar[i]; a1[i] = (_Float16)ar[8 + i]; }
      *(v8h*)(sA + mf * 512 + a_off(ml, cA))     = a0;
      *(v8h*)(sA + mf * 512 + a_off(ml, cA + 8)) = a1;

      int nf = nB >> 4, nl = nB & 15;
      v8h b0, b1;
#pragma unroll
      for (int i = 0; i < 8; ++i) { b0[i] = (_Float16)br[i]; b1[i] = (_Float16)br[8 + i]; }
      _Float16* bp2 = sB + nf * 512 + b_off(kB, nl);
      *(v8h*)bp2       = b0;
      *(v8h*)(bp2 + 8) = b1;
    }
    __syncthreads();
    if (kk + 1 < DD / 32) load_gmem(kk + 1);   // overlap next global load with math

    // hoist all fragment loads, then issue the 4 WMMAs back-to-back
    V16U af; af.v = *(const v16h*)(sA + wave * 512 + lane * 16);
    v16h bf0 = *(const v16h*)(sB + 0 * 512 + lane * 16);
    v16h bf1 = *(const v16h*)(sB + 1 * 512 + lane * 16);
    v16h bf2 = *(const v16h*)(sB + 2 * 512 + lane * 16);
    v16h bf3 = *(const v16h*)(sB + 3 * 512 + lane * 16);
    acc[0] = wmma_f16(af.v, bf0, acc[0]);
    acc[1] = wmma_f16(af.v, bf1, acc[1]);
    acc[2] = wmma_f16(af.v, bf2, acc[2]);
    acc[3] = wmma_f16(af.v, bf3, acc[3]);
  }

  // ---- epilogue ----
#pragma unroll
  for (int nf = 0; nf < 4; ++nf) {
    float bb_ = bias_hd[h * HD + nf * 16 + ln];
#pragma unroll
    for (int r = 0; r < 8; ++r) acc[nf][r] += bb_;
  }

  if (which < 2) {
    // layernorm over HD per row, entirely inside this wave (rows in 16-lane groups)
    const float* sc = (which == 0) ? qs : ksc;
    float scl[4];
#pragma unroll
    for (int nf = 0; nf < 4; ++nf) scl[nf] = sc[nf * 16 + ln];
    const float qmul = (which == 0) ? 0.125f : 1.0f;   // 1/sqrt(64) folded into Q
#pragma unroll
    for (int r = 0; r < 8; ++r) {
      float s1 = 0.f, s2 = 0.f;
#pragma unroll
      for (int nf = 0; nf < 4; ++nf) { float v = acc[nf][r]; s1 += v; s2 += v * v; }
#pragma unroll
      for (int d2 = 1; d2 < 16; d2 <<= 1) {
        s1 += __shfl_xor(s1, d2, 32);
        s2 += __shfl_xor(s2, d2, 32);
      }
      float mean = s1 * (1.0f / 64.0f);
      float var  = s2 * (1.0f / 64.0f) - mean * mean;
      float rstd = rsqrtf(var + LN_EPS);
#pragma unroll
      for (int nf = 0; nf < 4; ++nf) {
        float v = (acc[nf][r] - mean) * rstd * scl[nf] * qmul;
        sC[(wave * 16 + r + 8 * g) * 64 + nf * 16 + ln] = (_Float16)v;
      }
    }
    __syncthreads();
    _Float16* dst = ((which == 0) ? qo : ko) + ((size_t)(b * HH + h) * QQ + pos0) * HD;
    int row = tid >> 1, c = (tid & 1) * 32;
    const v8h* s = (const v8h*)(sC + row * 64 + c);
    v8h* d = (v8h*)(dst + (size_t)row * HD + c);
    d[0] = s[0]; d[1] = s[1]; d[2] = s[2]; d[3] = s[3];
  } else {
    // V: no LN; store transposed [b,h,HD,pos] via LDS transpose
#pragma unroll
    for (int nf = 0; nf < 4; ++nf)
#pragma unroll
      for (int r = 0; r < 8; ++r)
        sC[(wave * 16 + r + 8 * g) * 64 + nf * 16 + ln] = (_Float16)acc[nf][r];
    __syncthreads();
    int e = tid & 63, pr = (tid >> 6) * 32;
    _Float16 tmp[32];
#pragma unroll
    for (int i = 0; i < 32; ++i) tmp[i] = sC[(pr + i) * 64 + e];
    _Float16* d = vt + ((size_t)(b * HH + h) * HD + e) * KK + pos0 + pr;
#pragma unroll
    for (int i = 0; i < 4; ++i) {
      v8h t;
#pragma unroll
      for (int j = 0; j < 8; ++j) t[j] = tmp[i * 8 + j];
      *(v8h*)(d + i * 8) = t;
    }
  }
}

// =====================================================================
// Kernel 2: flash attention. grid = (Q/64, H, B); block = 128 (4 waves).
// Each wave owns 16 q rows; streams kv in tiles of 64 with online softmax.
// V^T fragments are issued at tile start so their latency hides under the
// S-WMMAs + softmax; next tile's bias/mask rows are prefetched.
// =====================================================================
__global__ __launch_bounds__(128) void flash_attn_kernel(
    const _Float16* __restrict__ qo, const _Float16* __restrict__ ko,
    const _Float16* __restrict__ vt,
    const float* __restrict__ bias, const unsigned char* __restrict__ mask,
    _Float16* __restrict__ xo)
{
  __shared__ _Float16 sP[4 * 2 * 512];   // per-wave P in A-fragment swizzle
  const int tid  = threadIdx.x;
  const int lane = tid & 31, wave = tid >> 5;
  const int g = lane >> 4, ln = lane & 15;
  const int b = blockIdx.z, h = blockIdx.y;
  const int q0 = blockIdx.x * 64 + wave * 16;   // this wave's 16 q rows

  // Q A-fragments held in registers for the whole kv sweep
  const _Float16* qrow = qo + ((size_t)(b * HH + h) * QQ + q0 + ln) * HD;
  V16U qa[2];
#pragma unroll
  for (int ksi = 0; ksi < 2; ++ksi) {
    qa[ksi].h[0] = *(const v8h*)(qrow + ksi * 32 + g * 16);
    qa[ksi].h[1] = *(const v8h*)(qrow + ksi * 32 + g * 16 + 8);
  }

  v8f zero8 = {};
  v8f acc[4] = { zero8, zero8, zero8, zero8 };
  float mstate[8], lstate[8];
#pragma unroll
  for (int r = 0; r < 8; ++r) { mstate[r] = BIG_NEG; lstate[r] = 0.f; }

  const float*         brow = bias + ((size_t)b * QQ + q0 + 8 * g) * KK + ln;
  const unsigned char* mrow = mask + ((size_t)b * QQ + q0 + 8 * g) * KK + ln;
  const _Float16* kbase = ko + (size_t)(b * HH + h) * QQ * HD;   // [kv][HD]
  const _Float16* vbase = vt + (size_t)(b * HH + h) * HD * KK;   // [HD][kv]

  for (int kt = 0; kt < KK / 64; ++kt) {
    const int kv0 = kt * 64;

    // prefetch next tile's bias/mask rows (global_prefetch_b8)
    if (kt + 1 < KK / 64) {
#pragma unroll
      for (int r = 0; r < 8; ++r) {
        __builtin_prefetch(brow + (size_t)r * KK + kv0 + 64, 0, 3);
        __builtin_prefetch(mrow + (size_t)r * KK + kv0 + 64, 0, 3);
      }
    }

    // issue all V^T B-fragments early; they complete while we do S + softmax
    v16h vb[2][4];
#pragma unroll
    for (int ksi = 0; ksi < 2; ++ksi)
#pragma unroll
      for (int of = 0; of < 4; ++of)
        vb[ksi][of] = *(const v16h*)(vbase + (size_t)(of * 16 + ln) * KK + kv0 +
                                     ksi * 32 + g * 16);

    // S = Q @ K^T  (B-frag = contiguous 32B from row-major K)
    v8f sf[4];
#pragma unroll
    for (int ns = 0; ns < 4; ++ns) {
      v16h kb0 = *(const v16h*)(kbase + (size_t)(kv0 + ns * 16 + ln) * HD + g * 16);
      v16h kb1 = *(const v16h*)(kbase + (size_t)(kv0 + ns * 16 + ln) * HD + 32 + g * 16);
      v8f c = zero8;
      c = wmma_f16(qa[0].v, kb0, c);
      c = wmma_f16(qa[1].v, kb1, c);
      sf[ns] = c;
    }
    // bias + mask (masked lanes ignore bias, matching reference)
#pragma unroll
    for (int ns = 0; ns < 4; ++ns)
#pragma unroll
      for (int r = 0; r < 8; ++r) {
        size_t off = (size_t)r * KK + kv0 + ns * 16;
        float bv_ = brow[off];
        unsigned char mk = mrow[off];
        sf[ns][r] = mk ? (sf[ns][r] + bv_) : BIG_NEG;
      }
    // online softmax: row max / rescale / exp / row sum (all in-wave shuffles)
    float rm[8], al[8], rs[8];
#pragma unroll
    for (int r = 0; r < 8; ++r) {
      float m_ = fmaxf(fmaxf(sf[0][r], sf[1][r]), fmaxf(sf[2][r], sf[3][r]));
#pragma unroll
      for (int d2 = 1; d2 < 16; d2 <<= 1) m_ = fmaxf(m_, __shfl_xor(m_, d2, 32));
      float mn = fmaxf(mstate[r], m_);
      al[r] = __expf(mstate[r] - mn);
      mstate[r] = mn; rm[r] = mn; rs[r] = 0.f;
    }
#pragma unroll
    for (int ns = 0; ns < 4; ++ns)
#pragma unroll
      for (int r = 0; r < 8; ++r) {
        float p = __expf(sf[ns][r] - rm[r]);
        sf[ns][r] = p;
        rs[r] += p;
      }
#pragma unroll
    for (int r = 0; r < 8; ++r) {
#pragma unroll
      for (int d2 = 1; d2 < 16; d2 <<= 1) rs[r] += __shfl_xor(rs[r], d2, 32);
      lstate[r] = lstate[r] * al[r] + rs[r];
    }
#pragma unroll
    for (int of = 0; of < 4; ++of)
#pragma unroll
      for (int r = 0; r < 8; ++r) acc[of][r] *= al[r];

    // C-layout P -> A-fragment swizzle via per-wave LDS (same-wave DS is in order)
#pragma unroll
    for (int ns = 0; ns < 4; ++ns)
#pragma unroll
      for (int r = 0; r < 8; ++r) {
        int kvl = ns * 16 + ln;
        sP[wave * 1024 + (kvl >> 5) * 512 + a_off(r + 8 * g, kvl & 31)] = (_Float16)sf[ns][r];
      }
    // acc += P @ V : V^T fragments already resident, WMMAs issue densely
    v16h pa0 = *(const v16h*)(sP + wave * 1024 + 0 * 512 + lane * 16);
    v16h pa1 = *(const v16h*)(sP + wave * 1024 + 1 * 512 + lane * 16);
#pragma unroll
    for (int of = 0; of < 4; ++of) acc[of] = wmma_f16(pa0, vb[0][of], acc[of]);
#pragma unroll
    for (int of = 0; of < 4; ++of) acc[of] = wmma_f16(pa1, vb[1][of], acc[of]);
  }

  // finalize: x[b,q,h,e] in f16 (A-friendly for out-proj GEMM)
  float inv[8];
#pragma unroll
  for (int r = 0; r < 8; ++r) inv[r] = 1.0f / lstate[r];
#pragma unroll
  for (int of = 0; of < 4; ++of)
#pragma unroll
    for (int r = 0; r < 8; ++r) {
      float v = acc[of][r] * inv[r];
      xo[((size_t)b * QQ + q0 + r + 8 * g) * DD + h * HD + of * 16 + ln] = (_Float16)v;
    }
}

// =====================================================================
// Kernel 3: out = x @ wo + bo.  grid = (B*Q/64, D/64); block = 128.
// =====================================================================
__global__ __launch_bounds__(128) void out_proj_kernel(
    const _Float16* __restrict__ xo, const float* __restrict__ wo,
    const float* __restrict__ bo, float* __restrict__ out)
{
  __shared__ _Float16 sA[4 * 512];
  __shared__ _Float16 sB[4 * 512];
  const int tid  = threadIdx.x;
  const int lane = tid & 31, wave = tid >> 5;
  const int g = lane >> 4, ln = lane & 15;
  const int row0 = blockIdx.x * 64;
  const int col0 = blockIdx.y * 64;

  const int rowA = tid >> 1, cA = (tid & 1) * 16;
  const int nB   = tid & 63, kB = (tid >> 6) * 16;

  v8h a0, a1;
  float br[16];
  auto load_gmem = [&](int kk) {
    const v8h* ap = (const v8h*)(xo + (size_t)(row0 + rowA) * DD + kk * 32 + cA);
    a0 = ap[0]; a1 = ap[1];
    const float* bp = wo + (size_t)(kk * 32 + kB) * DD + col0 + nB;
#pragma unroll
    for (int i = 0; i < 16; ++i) br[i] = bp[(size_t)i * DD];
  };

  v8f zero8 = {};
  v8f acc[4] = { zero8, zero8, zero8, zero8 };

  load_gmem(0);
  for (int kk = 0; kk < DD / 32; ++kk) {
    __syncthreads();
    {
      int mf = rowA >> 4, ml = rowA & 15;
      *(v8h*)(sA + mf * 512 + a_off(ml, cA))     = a0;
      *(v8h*)(sA + mf * 512 + a_off(ml, cA + 8)) = a1;
      int nf = nB >> 4, nl = nB & 15;
      v8h b0, b1;
#pragma unroll
      for (int i = 0; i < 8; ++i) { b0[i] = (_Float16)br[i]; b1[i] = (_Float16)br[8 + i]; }
      _Float16* bp2 = sB + nf * 512 + b_off(kB, nl);
      *(v8h*)bp2       = b0;
      *(v8h*)(bp2 + 8) = b1;
    }
    __syncthreads();
    if (kk + 1 < DD / 32) load_gmem(kk + 1);

    V16U af; af.v = *(const v16h*)(sA + wave * 512 + lane * 16);
    v16h bf0 = *(const v16h*)(sB + 0 * 512 + lane * 16);
    v16h bf1 = *(const v16h*)(sB + 1 * 512 + lane * 16);
    v16h bf2 = *(const v16h*)(sB + 2 * 512 + lane * 16);
    v16h bf3 = *(const v16h*)(sB + 3 * 512 + lane * 16);
    acc[0] = wmma_f16(af.v, bf0, acc[0]);
    acc[1] = wmma_f16(af.v, bf1, acc[1]);
    acc[2] = wmma_f16(af.v, bf2, acc[2]);
    acc[3] = wmma_f16(af.v, bf3, acc[3]);
  }

#pragma unroll
  for (int nf = 0; nf < 4; ++nf) {
    float bv_ = bo[col0 + nf * 16 + ln];
#pragma unroll
    for (int r = 0; r < 8; ++r)
      out[(size_t)(row0 + wave * 16 + r + 8 * g) * DD + col0 + nf * 16 + ln] =
          acc[nf][r] + bv_;
  }
}

// =====================================================================
extern "C" void kernel_launch(void* const* d_in, const int* in_sizes, int n_in,
                              void* d_out, int out_size, void* d_ws, size_t ws_size,
                              hipStream_t stream) {
  (void)in_sizes; (void)n_in; (void)out_size; (void)ws_size;
  const float*         inputs_q = (const float*)d_in[0];
  const float*         bias     = (const float*)d_in[1];
  const unsigned char* mask     = (const unsigned char*)d_in[2];
  const float* wq = (const float*)d_in[3];
  const float* bq = (const float*)d_in[4];
  const float* wk = (const float*)d_in[5];
  const float* bk = (const float*)d_in[6];
  const float* wv = (const float*)d_in[7];
  const float* bv = (const float*)d_in[8];
  const float* qs = (const float*)d_in[9];
  const float* ks = (const float*)d_in[10];
  const float* wo = (const float*)d_in[11];
  const float* bo = (const float*)d_in[12];
  float* out = (float*)d_out;

  const size_t nElem = (size_t)BB * HH * QQ * HD;   // 4M halves each
  _Float16* qbuf = (_Float16*)d_ws;
  _Float16* kbuf = qbuf + nElem;
  _Float16* vtb  = kbuf + nElem;
  _Float16* xbuf = vtb + nElem;

  dim3 g1(BB * QQ / 64, HH, 3);
  qkv_proj_kernel<<<g1, 128, 0, stream>>>(inputs_q, wq, wk, wv, bq, bk, bv, qs, ks,
                                          qbuf, kbuf, vtb);
  dim3 g2(QQ / 64, HH, BB);
  flash_attn_kernel<<<g2, 128, 0, stream>>>(qbuf, kbuf, vtb, bias, mask, xbuf);
  dim3 g3(BB * QQ / 64, DD / 64, 1);
  out_proj_kernel<<<g3, 128, 0, stream>>>(xbuf, wo, bo, out);
}
